// Branch1234_40458591928876
// MI455X (gfx1250) — compile-verified, matrix operand order checked by reading, not run on hardware
//
#include <hip/hip_runtime.h>
#include <hip/hip_bf16.h>
#include <math.h>

typedef __attribute__((ext_vector_type(16))) _Float16 v16h;
typedef __attribute__((ext_vector_type(8)))  _Float16 h8;
typedef __attribute__((ext_vector_type(8)))  float    v8f;
typedef __attribute__((ext_vector_type(4)))  float    f4;

#define TILE 32

// Gather one WMMA operand fragment from an LDS row laid out [.][32+8 pad].
// Per the CDNA5 16-bit A/B layout, lane (grp = lane>>4) needs K runs
// [8*grp, 8*grp+8) and [16+8*grp, 16+8*grp+8): two aligned 16B LDS loads.
__device__ __forceinline__ v16h frag_load(const _Float16* rowp, int grp)
{
    h8 lo = *(const h8*)(rowp + grp * 8);
    h8 hi = *(const h8*)(rowp + 16 + grp * 8);
    v16h f;
    #pragma unroll
    for (int j = 0; j < 8; ++j) { f[j] = lo[j]; f[8 + j] = hi[j]; }
    return f;
}

// ---------------------------------------------------------------------------
// Aligned WMMA GEMM: requires M%64==0, N%64==0, K%32==0, lda%4==0, ldb%4==0.
// Block = 128 threads (4 waves). Block tile 64x64; each wave owns a 32x32
// sub-tile -> 4 WMMAs per K-step (2 A-frags x 2 B-frags). No bounds checks:
// b128 global loads, packed b128 LDS stores for A, f16 convert in staging.
// ---------------------------------------------------------------------------
__global__ __launch_bounds__(128)
void wmma_gemm_aligned(const float* __restrict__ A, int lda,
                       const float* __restrict__ B, int ldb,
                       float* __restrict__ C, int ldc,
                       const float* __restrict__ bias,
                       int M, int N, int K, int accumulate)
{
    __shared__ _Float16 As[64][TILE + 8];   // [m][k]
    __shared__ _Float16 Bs[64][TILE + 8];   // [n][k] (transposed stage)

    const int tid  = threadIdx.x;
    const int lane = tid & 31;
    const int wave = tid >> 5;
    const int wr   = wave >> 1;             // 0/1 : row half of block tile
    const int wc   = wave & 1;              // 0/1 : col half of block tile
    const int m0b  = blockIdx.y * 64;
    const int n0b  = blockIdx.x * 64;
    const int grp  = lane >> 4;
    const int hrow = lane & 15;

    // staging assignments
    const int ar  = tid >> 1;               // A: row 0..63
    const int ac0 = (tid & 1) * 16;         // A: col 0 or 16
    const int bk  = tid >> 2;               // B: k row 0..31
    const int bc0 = (tid & 3) * 16;         // B: n col 0/16/32/48

    v8f acc00 = {}, acc01 = {}, acc10 = {}, acc11 = {};

    const int ksteps = K >> 5;
    for (int kt = 0; kt < ksteps; ++kt) {
        const int k0 = kt * TILE;

        // ---- stage A: 64x32 floats, 16 per thread via 4x b128 loads ----
        {
            const float* pA = A + (size_t)(m0b + ar) * lda + k0 + ac0;
            f4 a0 = *(const f4*)(pA);
            f4 a1 = *(const f4*)(pA + 4);
            f4 a2 = *(const f4*)(pA + 8);
            f4 a3 = *(const f4*)(pA + 12);
            h8 h0, h1;
            #pragma unroll
            for (int j = 0; j < 4; ++j) {
                h0[j]     = (_Float16)a0[j];
                h0[4 + j] = (_Float16)a1[j];
                h1[j]     = (_Float16)a2[j];
                h1[4 + j] = (_Float16)a3[j];
            }
            *(h8*)&As[ar][ac0]     = h0;
            *(h8*)&As[ar][ac0 + 8] = h1;
        }
        // ---- stage B transposed: 32x64 floats -> Bs[n][k] ----
        {
            const float* pB = B + (size_t)(k0 + bk) * ldb + n0b + bc0;
            f4 b0 = *(const f4*)(pB);
            f4 b1 = *(const f4*)(pB + 4);
            f4 b2 = *(const f4*)(pB + 8);
            f4 b3 = *(const f4*)(pB + 12);
            #pragma unroll
            for (int j = 0; j < 4; ++j) {
                Bs[bc0 + j][bk]      = (_Float16)b0[j];
                Bs[bc0 + 4 + j][bk]  = (_Float16)b1[j];
                Bs[bc0 + 8 + j][bk]  = (_Float16)b2[j];
                Bs[bc0 + 12 + j][bk] = (_Float16)b3[j];
            }
        }
        // prefetch next K tile (global_prefetch_b8)
        if (kt + 1 < ksteps) {
            __builtin_prefetch(A + (size_t)(m0b + ar) * lda + k0 + TILE + ac0, 0, 1);
            __builtin_prefetch(B + (size_t)(k0 + TILE + bk) * ldb + n0b + bc0, 0, 1);
        }
        __syncthreads();

        v16h a0 = frag_load(&As[wr * 32 + hrow][0], grp);
        v16h a1 = frag_load(&As[wr * 32 + 16 + hrow][0], grp);
        v16h b0 = frag_load(&Bs[wc * 32 + hrow][0], grp);
        v16h b1 = frag_load(&Bs[wc * 32 + 16 + hrow][0], grp);

        acc00 = __builtin_amdgcn_wmma_f32_16x16x32_f16(false, a0, false, b0, (short)0, acc00, false, false);
        acc01 = __builtin_amdgcn_wmma_f32_16x16x32_f16(false, a0, false, b1, (short)0, acc01, false, false);
        acc10 = __builtin_amdgcn_wmma_f32_16x16x32_f16(false, a1, false, b0, (short)0, acc10, false, false);
        acc11 = __builtin_amdgcn_wmma_f32_16x16x32_f16(false, a1, false, b1, (short)0, acc11, false, false);
        __syncthreads();
    }

    // writeback (no guards; everything aligned)
    const int n0 = n0b + wc * 32 + hrow;
    const int n1 = n0 + 16;
    const int m0 = m0b + wr * 32 + grp * 8;
    const int m1 = m0 + 16;
    const float bias0 = bias ? bias[n0] : 0.0f;
    const float bias1 = bias ? bias[n1] : 0.0f;
    #pragma unroll
    for (int r = 0; r < 8; ++r) {
        size_t o00 = (size_t)(m0 + r) * ldc + n0;
        size_t o01 = (size_t)(m0 + r) * ldc + n1;
        size_t o10 = (size_t)(m1 + r) * ldc + n0;
        size_t o11 = (size_t)(m1 + r) * ldc + n1;
        if (accumulate) {
            C[o00] += acc00[r]; C[o01] += acc01[r];
            C[o10] += acc10[r]; C[o11] += acc11[r];
        } else {
            C[o00] = acc00[r] + bias0; C[o01] = acc01[r] + bias1;
            C[o10] = acc10[r] + bias0; C[o11] = acc11[r] + bias1;
        }
    }
}

// ---------------------------------------------------------------------------
// Guarded WMMA GEMM for ragged shapes (N=40, K=8, N=96 here). Branchless
// staging: indices clamped so loads are always in-bounds, invalid elements
// zeroed via select (v_cndmask) -- no per-element exec-mask dances.
// Block = 128 threads, 32x32 block tile, one 16x16 WMMA tile per wave.
// ---------------------------------------------------------------------------
__global__ __launch_bounds__(128)
void wmma_gemm_guarded(const float* __restrict__ A, int lda,
                       const float* __restrict__ B, int ldb,
                       float* __restrict__ C, int ldc,
                       const float* __restrict__ bias,
                       int M, int N, int K, int accumulate)
{
    __shared__ _Float16 As[TILE][TILE + 8];
    __shared__ _Float16 Bs[TILE][TILE + 8];   // [n][k]

    const int tid  = threadIdx.x;
    const int lane = tid & 31;
    const int wave = tid >> 5;
    const int wr   = wave >> 1;
    const int wc   = wave & 1;
    const int m0b  = blockIdx.y * TILE;
    const int n0b  = blockIdx.x * TILE;
    const int grp  = lane >> 4;
    const int hrow = lane & 15;

    v8f acc = {};

    const int ksteps = (K + TILE - 1) / TILE;
    for (int kt = 0; kt < ksteps; ++kt) {
        const int k0 = kt * TILE;
        #pragma unroll
        for (int i = tid; i < TILE * TILE; i += 128) {
            int r = i >> 5, c = i & 31;
            int gm = m0b + r, gk = k0 + c;
            int cm = gm < M ? gm : M - 1;
            int ck = gk < K ? gk : K - 1;
            float v = A[(size_t)cm * lda + ck];
            As[r][c] = (_Float16)((gm < M && gk < K) ? v : 0.0f);
        }
        #pragma unroll
        for (int i = tid; i < TILE * TILE; i += 128) {
            int r = i >> 5, c = i & 31;          // r = k, c = n
            int gk = k0 + r, gn = n0b + c;
            int ck = gk < K ? gk : K - 1;
            int cn = gn < N ? gn : N - 1;
            float v = B[(size_t)ck * ldb + cn];
            Bs[c][r] = (_Float16)((gk < K && gn < N) ? v : 0.0f);
        }
        __syncthreads();

        v16h af = frag_load(&As[wr * 16 + hrow][0], grp);
        v16h bf = frag_load(&Bs[wc * 16 + hrow][0], grp);
        acc = __builtin_amdgcn_wmma_f32_16x16x32_f16(false, af, false, bf, (short)0, acc, false, false);
        __syncthreads();
    }

    const int n     = n0b + wc * 16 + hrow;
    const int mbase = m0b + wr * 16 + grp * 8;
    #pragma unroll
    for (int r = 0; r < 8; ++r) {
        int m = mbase + r;
        if (m < M && n < N) {
            size_t off = (size_t)m * ldc + n;
            float v = acc[r];
            if (accumulate) C[off] += v;
            else            C[off]  = v + (bias ? bias[n] : 0.0f);
        }
    }
}

// ---------------------------------------------------------------------------
// x:(b,c,64,64) -> tokens t:(b, l=h*64+w, c)
// ---------------------------------------------------------------------------
__global__ __launch_bounds__(256)
void tokens_kernel(const float* __restrict__ x, float* __restrict__ t)
{
    int i = blockIdx.x * 256 + threadIdx.x;           // i = ((b*128+c)*4096)+l
    if (i >= 4 * 128 * 4096) return;
    int l = i & 4095;
    int c = (i >> 12) & 127;
    int b = i >> 19;
    t[((size_t)b * 4096 + l) * 128 + c] = x[i];
}

// ---------------------------------------------------------------------------
// Causal depthwise conv (kw=4) + bias + SiLU.
// ---------------------------------------------------------------------------
__global__ __launch_bounds__(256)
void conv_silu_kernel(const float* __restrict__ xz, int ldxz,
                      const float* __restrict__ convw,
                      const float* __restrict__ convb,
                      float* __restrict__ xp,
                      int L, int di, long total)
{
    long i = (long)blockIdx.x * 256 + threadIdx.x;
    if (i >= total) return;
    int  d   = (int)(i % di);
    long row = i / di;                 // b*L + l
    int  l   = (int)(row % L);
    long b   = row / L;
    float acc = convb[d];
    #pragma unroll
    for (int k = 0; k < 4; ++k) {
        int li = l - 3 + k;
        if (li >= 0)
            acc += xz[((size_t)(b * L + li)) * ldxz + d] * convw[d * 4 + k];
    }
    xp[(size_t)row * di + d] = acc / (1.0f + expf(-acc));   // silu
}

// ---------------------------------------------------------------------------
// Selective scan: one (b,d) per thread, 16 states in registers,
// B/C rows broadcast via LDS. Applies softplus(dt), +x*D, gates by silu(z).
// ---------------------------------------------------------------------------
__global__ __launch_bounds__(256)
void scan_kernel(const float* __restrict__ xp,
                 const float* __restrict__ dtp,
                 const float* __restrict__ dbl, int ldbl, int boff,
                 const float* __restrict__ xz, int ldxz,
                 const float* __restrict__ Alog,
                 const float* __restrict__ Dp,
                 float* __restrict__ ybuf,
                 int L, int di)
{
    __shared__ float sB[16];
    __shared__ float sC[16];
    const int tid = threadIdx.x;
    const int dgroups = di >> 8;
    const int b  = blockIdx.x / dgroups;
    const int dg = blockIdx.x % dgroups;
    const int d  = dg * 256 + tid;

    float Areg[16], h[16];
    #pragma unroll
    for (int s = 0; s < 16; ++s) {
        Areg[s] = -expf(Alog[(size_t)d * 16 + s]);
        h[s] = 0.0f;
    }
    const float Dd = Dp[d];

    for (int l = 0; l < L; ++l) {
        size_t row = (size_t)b * L + l;
        if (tid < 16)       sB[tid]      = dbl[row * ldbl + boff + tid];
        else if (tid < 32)  sC[tid - 16] = dbl[row * ldbl + boff + 16 + (tid - 16)];
        __syncthreads();

        float dtr = dtp[row * di + d];
        float dtv = (dtr > 20.0f) ? dtr : log1pf(expf(dtr));
        float xv  = xp[row * di + d];
        float dx  = dtv * xv;
        float y   = 0.0f;
        #pragma unroll
        for (int s = 0; s < 16; ++s) {
            float dA = expf(dtv * Areg[s]);
            h[s] = dA * h[s] + dx * sB[s];
            y   += h[s] * sC[s];
        }
        y += xv * Dd;
        float zv = xz[row * ldxz + di + d];
        y *= zv / (1.0f + expf(-zv));            // * silu(z)
        ybuf[row * di + d] = y;
        __syncthreads();
    }
}

// ---------------------------------------------------------------------------
// Bilinear downsample x:(b,c,64,64) -> s3:(b*c, 32*32)
// ---------------------------------------------------------------------------
__global__ __launch_bounds__(256)
void bilin_down_kernel(const float* __restrict__ x, float* __restrict__ s3)
{
    int i = blockIdx.x * 256 + threadIdx.x;
    if (i >= 4 * 128 * 32 * 32) return;
    int q  = i & 31;
    int p  = (i >> 5) & 31;
    int bc = i >> 10;
    const float scale = 63.0f / 31.0f;
    float ph = p * scale; int lp = (int)ph; float wp = ph - lp;
    int hp = lp + 1; if (hp > 63) hp = 63;
    float pq = q * scale; int lq = (int)pq; float wq = pq - lq;
    int hq = lq + 1; if (hq > 63) hq = 63;
    const float* xb = x + (size_t)bc * 4096;
    float v = (1.0f - wp) * ((1.0f - wq) * xb[lp * 64 + lq] + wq * xb[lp * 64 + hq])
            +         wp  * ((1.0f - wq) * xb[hp * 64 + lq] + wq * xb[hp * 64 + hq]);
    s3[(size_t)bc * 1024 + p * 32 + q] = v;
}

// ---------------------------------------------------------------------------
// out(b,c,h,w) = acc_t(b,l,c) + up32->64(g3) + up32->64(g4)
// ---------------------------------------------------------------------------
__global__ __launch_bounds__(256)
void combine_kernel(const float* __restrict__ acc_t,
                    const float* __restrict__ g3,
                    const float* __restrict__ g4,
                    float* __restrict__ out)
{
    int i = blockIdx.x * 256 + threadIdx.x;
    if (i >= 4 * 128 * 64 * 64) return;
    int w  = i & 63;
    int h  = (i >> 6) & 63;
    int c  = (i >> 12) & 127;
    int b  = i >> 19;
    int l  = h * 64 + w;
    float v = acc_t[((size_t)b * 4096 + l) * 128 + c];

    const float sc = 31.0f / 63.0f;
    float ph = h * sc; int lh = (int)ph; float wh = ph - lh;
    int hh = lh + 1; if (hh > 31) hh = 31;
    float pw = w * sc; int lw = (int)pw; float ww = pw - lw;
    int hw = lw + 1; if (hw > 31) hw = 31;

    size_t base = (size_t)(b * 128 + c) * 1024;
    float a00 = (1.0f - wh) * (1.0f - ww), a01 = (1.0f - wh) * ww;
    float a10 = wh * (1.0f - ww),          a11 = wh * ww;
    int i00 = lh * 32 + lw, i01 = lh * 32 + hw;
    int i10 = hh * 32 + lw, i11 = hh * 32 + hw;

    v += a00 * g3[base + i00] + a01 * g3[base + i01]
       + a10 * g3[base + i10] + a11 * g3[base + i11];
    v += a00 * g4[base + i00] + a01 * g4[base + i01]
       + a10 * g4[base + i10] + a11 * g4[base + i11];
    out[i] = v;
}

// ---------------------------------------------------------------------------
extern "C" void kernel_launch(void* const* d_in, const int* in_sizes, int n_in,
                              void* d_out, int out_size, void* d_ws, size_t ws_size,
                              hipStream_t stream)
{
    (void)in_sizes; (void)n_in; (void)out_size; (void)ws_size;
    const float* x        = (const float*)d_in[0];
    const float* W_ch     = (const float*)d_in[1];
    const float* b_ch     = (const float*)d_in[2];
    const float* W_spa    = (const float*)d_in[3];
    const float* b_spa    = (const float*)d_in[4];
    const float* mc_Win   = (const float*)d_in[5];
    const float* mc_convw = (const float*)d_in[6];
    const float* mc_convb = (const float*)d_in[7];
    const float* mc_Wx    = (const float*)d_in[8];
    const float* mc_Wdt   = (const float*)d_in[9];
    const float* mc_bdt   = (const float*)d_in[10];
    const float* mc_Alog  = (const float*)d_in[11];
    const float* mc_D     = (const float*)d_in[12];
    const float* mc_Wout  = (const float*)d_in[13];
    const float* ms_Win   = (const float*)d_in[14];
    const float* ms_convw = (const float*)d_in[15];
    const float* ms_convb = (const float*)d_in[16];
    const float* ms_Wx    = (const float*)d_in[17];
    const float* ms_Wdt   = (const float*)d_in[18];
    const float* ms_bdt   = (const float*)d_in[19];
    const float* ms_Alog  = (const float*)d_in[20];
    const float* ms_D     = (const float*)d_in[21];
    const float* ms_Wout  = (const float*)d_in[22];

    float* ws = (float*)d_ws;
    size_t off = 0;
    auto alloc = [&](size_t n) { float* p = ws + off; off += n; return p; };
    float* t     = alloc((size_t)16384 * 128);
    float* acc_t = alloc((size_t)16384 * 128);
    float* xz_c  = alloc((size_t)16384 * 512);
    float* xp_c  = alloc((size_t)16384 * 256);
    float* dbl_c = alloc((size_t)16384 * 40);
    float* dtp_c = alloc((size_t)16384 * 256);
    float* y_c   = alloc((size_t)16384 * 256);
    float* s3    = alloc((size_t)512 * 1024);
    float* xz_s  = alloc((size_t)512 * 4096);
    float* xp_s  = alloc((size_t)512 * 2048);
    float* dbl_s = alloc((size_t)512 * 96);
    float* dtp_s = alloc((size_t)512 * 2048);
    float* y_s   = alloc((size_t)512 * 2048);
    float* g3    = alloc((size_t)512 * 1024);
    float* g4    = alloc((size_t)512 * 1024);

    auto gemm = [&](const float* A, int lda, const float* B, int ldb,
                    float* C, int ldc, const float* bias,
                    int M, int N, int K, int accFlag) {
        bool aligned = (M % 64 == 0) && (N % 64 == 0) && (K % 32 == 0) &&
                       (lda % 4 == 0) && (ldb % 4 == 0);
        if (aligned) {
            dim3 grid(N / 64, M / 64);
            wmma_gemm_aligned<<<grid, dim3(128), 0, stream>>>(
                A, lda, B, ldb, C, ldc, bias, M, N, K, accFlag);
        } else {
            dim3 grid((N + TILE - 1) / TILE, (M + TILE - 1) / TILE);
            wmma_gemm_guarded<<<grid, dim3(128), 0, stream>>>(
                A, lda, B, ldb, C, ldc, bias, M, N, K, accFlag);
        }
    };

    // tokens
    tokens_kernel<<<(4 * 128 * 4096 + 255) / 256, 256, 0, stream>>>(x, t);

    // Branch 1: acc_t = t @ W_ch + b_ch
    gemm(t, 128, W_ch, 128, acc_t, 128, b_ch, 16384, 128, 128, 0);

    // Branch 2: mamba over tokens (L=4096, dm=128, di=256, dtr=8)
    gemm(t, 128, mc_Win, 512, xz_c, 512, nullptr, 16384, 512, 128, 0);
    conv_silu_kernel<<<((long)16384 * 256 + 255) / 256, 256, 0, stream>>>(
        xz_c, 512, mc_convw, mc_convb, xp_c, 4096, 256, (long)16384 * 256);
    gemm(xp_c, 256, mc_Wx, 40, dbl_c, 40, nullptr, 16384, 40, 256, 0);       // guarded
    gemm(dbl_c, 40, mc_Wdt, 256, dtp_c, 256, mc_bdt, 16384, 256, 8, 0);      // guarded
    scan_kernel<<<4, 256, 0, stream>>>(xp_c, dtp_c, dbl_c, 40, 8,
                                       xz_c, 512, mc_Alog, mc_D, y_c, 4096, 256);
    gemm(y_c, 256, mc_Wout, 128, acc_t, 128, nullptr, 16384, 128, 256, 1);   // +=

    // Branch 3/4 shared: s3 = bilinear(x, 32, 32)
    bilin_down_kernel<<<(4 * 128 * 32 * 32 + 255) / 256, 256, 0, stream>>>(x, s3);

    // Branch 3: mamba over channels (L=128, dm=1024, di=2048, dtr=64)
    gemm(s3, 1024, ms_Win, 4096, xz_s, 4096, nullptr, 512, 4096, 1024, 0);
    conv_silu_kernel<<<((long)512 * 2048 + 255) / 256, 256, 0, stream>>>(
        xz_s, 4096, ms_convw, ms_convb, xp_s, 128, 2048, (long)512 * 2048);
    gemm(xp_s, 2048, ms_Wx, 96, dbl_s, 96, nullptr, 512, 96, 2048, 0);       // guarded (N=96)
    gemm(dbl_s, 96, ms_Wdt, 2048, dtp_s, 2048, ms_bdt, 512, 2048, 64, 0);
    scan_kernel<<<4 * 8, 256, 0, stream>>>(xp_s, dtp_s, dbl_s, 96, 64,
                                           xz_s, 4096, ms_Alog, ms_D, y_s, 128, 2048);
    gemm(y_s, 2048, ms_Wout, 1024, g3, 1024, nullptr, 512, 1024, 2048, 0);

    // Branch 4: g4 = s3 @ W_spa + b_spa
    gemm(s3, 1024, W_spa, 1024, g4, 1024, b_spa, 512, 1024, 1024, 0);

    // out = branch1+2 (token layout) + up(g3) + up(g4)
    combine_kernel<<<(4 * 128 * 64 * 64 + 255) / 256, 256, 0, stream>>>(
        acc_t, g3, g4, (float*)d_out);
}
